// SchrodingerAttention_25769803776308
// MI455X (gfx1250) — compile-verified, matrix operand order checked by reading, not run on hardware
//
#include <hip/hip_runtime.h>
#include <hip/hip_bf16.h>

// ---------------- problem constants (from reference) ----------------
#define BB   2
#define LL   4096
#define DD   1024
#define SD_  1024
#define SC_  1024
#define NB_  4
#define PP   8
#define NROW (BB * LL)          // 8192 rows of x
#define EPSF 1e-6f

// GEMM tiling: 4 waves (128 thr) per block, block tile 128x128, wave tile 64x64.
#define BM 128
#define BN 128
#define BK 32

typedef __bf16 bf16;
typedef __attribute__((ext_vector_type(16))) __bf16 v16bf;
typedef __attribute__((ext_vector_type(8)))  __bf16 v8bf;
typedef __attribute__((ext_vector_type(8)))  float  v8f;

__device__ __forceinline__ float softplusf(float x) {
  return (x > 20.f) ? x : log1pf(__expf(x));
}

// ISA 10.2: flat addresses in the LDS aperture map to LDS via addr[31:0],
// so the low 32 bits of a generic pointer to __shared__ ARE the LDS byte offset.
__device__ __forceinline__ unsigned lds_off(const void* p) {
  return (unsigned)(unsigned long long)p;
}

// CDNA5 async copy: memory -> LDS, no VGPR staging, tracked by ASYNCcnt.
__device__ __forceinline__ void async_copy_b128(unsigned lds, const void* gptr) {
  asm volatile("global_load_async_to_lds_b128 %0, %1, off"
               :: "v"(lds), "v"((unsigned long long)gptr)
               : "memory");
}
__device__ __forceinline__ void wait_asynccnt0() {
  asm volatile("s_wait_asynccnt 0x0" ::: "memory");
}

// ---------------- f32 -> bf16 pack ----------------
__global__ void cvt_bf16_kernel(const float* __restrict__ in, bf16* __restrict__ out, size_t n) {
  size_t i = (size_t)blockIdx.x * blockDim.x + threadIdx.x;
  if (i < n) out[i] = (bf16)in[i];
}

// ---------------- WMMA bf16 GEMM: C[M,N] = A[M,K] @ W[N,K]^T + bias (+ addC) ----------------
// Async-LDS staged: each block pulls a 128x32 A-slab and 128x32 W-slab into LDS
// via global_load_async_to_lds_b128; 4 waves each compute a 64x64 tile as
// 4x4 v_wmma_f32_16x16x32_bf16 fragments fed by ds_load_b128.

// Load a 16x32 bf16 fragment (A-layout) from an LDS slab stored [row][BK].
__device__ __forceinline__ v16bf lds_frag16(const bf16* __restrict__ s, int row0, int lane) {
  const int m  = lane & 15;
  const int kh = (lane >> 4) << 3;                    // 0 or 8
  const bf16* p = s + (size_t)(row0 + m) * BK + kh;
  v8bf lo = *(const v8bf*)(p);                        // K = kh    .. +7
  v8bf hi = *(const v8bf*)(p + 16);                   // K = 16+kh .. +7
  v16bf r;
#pragma unroll
  for (int i = 0; i < 8; ++i) { r[i] = lo[i]; r[i + 8] = hi[i]; }
  return r;
}

__global__ __launch_bounds__(128) void wmma_gemm_bias(
    const bf16* __restrict__ A, const bf16* __restrict__ W,
    const float* __restrict__ bias, const float* __restrict__ addC,
    float* __restrict__ C, int M, int N, int K) {
  __shared__ bf16 As[BM * BK];                        // 8 KB
  __shared__ bf16 Bs[BN * BK];                        // 8 KB

  const int tid  = threadIdx.x;
  const int lane = tid & 31;
  const int wave = tid >> 5;                          // 0..3
  const int wm   = (wave >> 1) * 64;                  // wave row offset in block tile
  const int wn   = (wave & 1) * 64;                   // wave col offset in block tile
  const int n0 = blockIdx.x * BN;
  const int m0 = blockIdx.y * BM;

  v8f acc[4][4] = {};

  for (int k = 0; k < K; k += BK) {
    // ---- async fill: 512 16B chunks each for As/Bs, 4 per thread, no VGPR staging ----
#pragma unroll
    for (int i = 0; i < 4; ++i) {
      const int c    = tid + i * 128;                 // chunk id 0..511
      const int row  = c >> 2;                        // 4 chunks (64B) per row
      const int koff = (c & 3) << 3;                  // 0,8,16,24
      async_copy_b128(lds_off(As + (size_t)row * BK + koff),
                      A + (size_t)(m0 + row) * K + k + koff);
      async_copy_b128(lds_off(Bs + (size_t)row * BK + koff),
                      W + (size_t)(n0 + row) * K + k + koff);
    }
    if (k + BK < K) {                                 // prefetch next K-slab (global_prefetch_b8)
      __builtin_prefetch((const void*)(A + (size_t)(m0 + tid) * K + k + BK), 0, 1);
      __builtin_prefetch((const void*)(W + (size_t)(n0 + tid) * K + k + BK), 0, 1);
    }
    wait_asynccnt0();                                 // this wave's copies landed in LDS
    __syncthreads();                                  // everyone's copies visible

    v16bf af[4], bfm[4];
#pragma unroll
    for (int i = 0; i < 4; ++i) af[i]  = lds_frag16(As, wm + 16 * i, lane);
#pragma unroll
    for (int j = 0; j < 4; ++j) bfm[j] = lds_frag16(Bs, wn + 16 * j, lane);
#pragma unroll
    for (int i = 0; i < 4; ++i)
#pragma unroll
      for (int j = 0; j < 4; ++j)
        acc[i][j] = __builtin_amdgcn_wmma_f32_16x16x32_bf16(
            false, af[i], false, bfm[j], (short)0, acc[i][j], false, false);
    __syncthreads();                                  // protect LDS before next fill
  }

  // C layout: lane<16 -> N=lane,    M = vgpr r     (r=0..7)
  //           lane>=16 -> N=lane-16, M = vgpr r + 8
  const int col = lane & 15;
  const int rb  = (lane >> 4) << 3;
#pragma unroll
  for (int i = 0; i < 4; ++i)
#pragma unroll
    for (int j = 0; j < 4; ++j) {
      const int n = n0 + wn + 16 * j + col;
      const float bv = bias ? bias[n] : 0.f;
#pragma unroll
      for (int r = 0; r < 8; ++r) {
        const size_t m = (size_t)(m0 + wm + 16 * i + rb + r);
        float v = acc[i][j][r] + bv;
        if (addC) v += addC[m * (size_t)N + n];
        C[m * (size_t)N + n] = v;
      }
    }
}

// ---------------- gate = sigmoid((x.Wg + bg)/softplus(tau) + beta) ----------------
__global__ void gate_kernel(const float* __restrict__ x, const float* __restrict__ Wg,
                            const float* __restrict__ bg, const float* __restrict__ tau,
                            const float* __restrict__ beta, float* __restrict__ gate) {
  int n = blockIdx.x * blockDim.x + threadIdx.x;
  if (n >= NROW) return;
  const float* row = x + (size_t)n * DD;
  float s = 0.f;
  for (int k = 0; k < DD; ++k) s += row[k] * Wg[k];
  float z = (s + bg[0]) / softplusf(tau[0]) + beta[0];
  gate[n] = 1.f / (1.f + __expf(-z));
}

// ---------------- U = (softplus(Braw)*gate) * complex(psi) * phasor(ph) ----------------
// Works for both scout (SC) and true (SD) waves since SC_ == SD_.
__global__ void u_kernel(const float* __restrict__ psi_raw,   // [N, 2*Dc]
                         const float* __restrict__ ph_raw,    // [N, 2*Dc]
                         const float* __restrict__ B_raw,     // [N, Dc] (pre-softplus)
                         const float* __restrict__ gate,      // [N]
                         float2* __restrict__ U) {             // [N, Dc] interleaved
  size_t idx = (size_t)blockIdx.x * blockDim.x + threadIdx.x;
  if (idx >= (size_t)NROW * SD_) return;
  int n = (int)(idx / SD_), d = (int)(idx % SD_);
  const size_t b2 = (size_t)n * 2 * SD_;
  float pre = psi_raw[b2 + d];
  float pim = psi_raw[b2 + SD_ + d];
  float pa = ph_raw[b2 + 2 * d];
  float pb = ph_raw[b2 + 2 * d + 1];
  float inv = rsqrtf(fmaxf(pa * pa + pb * pb, 1e-24f));
  float phr = pa * inv, phm = pb * inv;
  float s = softplusf(B_raw[idx]) * gate[n];
  U[idx] = make_float2(s * (pre * phr - pim * phm), s * (pre * phm + pim * phr));
}

// ---------------- scout scan: h_t = a h_{t-1} + u_t ; feat[t] = h_{t-1} (bf16) ----------------
__global__ void scan_scout_kernel(const float2* __restrict__ U,       // [B,L,SC]
                                  const float* __restrict__ omega, const float* __restrict__ log_gamma,
                                  const float* __restrict__ dt_raw,
                                  bf16* __restrict__ feat) {           // [B,L,2*SC]
  int idx = blockIdx.x * blockDim.x + threadIdx.x;
  if (idx >= BB * SC_) return;
  int b = idx / SC_, c = idx % SC_;
  float dt = softplusf(dt_raw[0]);
  float mask = (c >= SC_ - SC_ / NB_) ? 1.f : 0.f;    // lowest-freq band = largest n
  float g = softplusf(log_gamma[c]) * (1.f - mask);
  float mag = __expf(-g * dt);
  float th = omega[c] * dt;
  float ar = mag * __cosf(th), ai = mag * __sinf(th);
  const float2* u = U + (size_t)b * LL * SC_ + c;
  bf16* f = feat + (size_t)b * LL * 2 * SC_;
  float hr = 0.f, hi = 0.f;
  for (int t = 0; t < LL; ++t) {
    f[(size_t)t * 2 * SC_ + c]        = (bf16)hr;     // shifted: feat[t] = h_{t-1}
    f[(size_t)t * 2 * SC_ + SC_ + c]  = (bf16)hi;
    float2 uv = u[(size_t)t * SC_];
    float nr = ar * hr - ai * hi + uv.x;
    float ni = ar * hi + ai * hr + uv.y;
    hr = nr; hi = ni;
  }
}

// ---------------- true scan: Simpson + recurrence + fractal map (in-place on U) ----------------
__global__ void scan_true_kernel(float2* __restrict__ U,              // in: U, out: mapped h
                                 const float* __restrict__ omega, const float* __restrict__ log_gamma,
                                 const float* __restrict__ dt_raw,
                                 const float* __restrict__ fa,        // [SD,2]
                                 const float* __restrict__ fbr, const float* __restrict__ fbi,
                                 const float* __restrict__ fc,  const float* __restrict__ fd) {
  int idx = blockIdx.x * blockDim.x + threadIdx.x;
  if (idx >= BB * SD_) return;
  int b = idx / SD_, d = idx % SD_;
  float dt = softplusf(dt_raw[0]);
  float mask = (d >= SD_ - SD_ / NB_) ? 1.f : 0.f;
  float g = softplusf(log_gamma[d]) * (1.f - mask);
  float mag = __expf(-g * dt);
  float th = omega[d] * dt;
  float ar = mag * __cosf(th), ai = mag * __sinf(th);
  // fractal constants
  float ax = fa[2 * d], ay = fa[2 * d + 1];
  float inv = rsqrtf(fmaxf(ax * ax + ay * ay, 1e-24f));
  ax *= inv; ay *= inv;
  float br = fbr[d], bi = fbi[d];
  float cc = softplusf(fc[d]);
  float ddc = softplusf(fd[d]) + EPSF;
  float2* p = U + (size_t)b * LL * SD_ + d;
  float u1r = 0.f, u1i = 0.f, u2r = 0.f, u2i = 0.f, hr = 0.f, hi = 0.f;
  const float c6 = 1.f / 6.f;
  for (int t = 0; t < LL; ++t) {
    float2 uv = p[(size_t)t * SD_];
    float sr = (uv.x + 4.f * u1r + u2r) * c6;         // Simpson [1,4,1]/6 causal
    float si = (uv.y + 4.f * u1i + u2i) * c6;
    u2r = u1r; u2i = u1i; u1r = uv.x; u1i = uv.y;
    float nr = ar * hr - ai * hi + sr;
    float ni = ar * hi + ai * hr + si;
    hr = nr; hi = ni;
    float m = sqrtf(hr * hr + hi * hi);
    float den = 1.f / (cc * m + ddc);
    p[(size_t)t * SD_] = make_float2((ax * hr - ay * hi + br) * den,
                                     (ax * hi + ay * hr + bi) * den);
  }
}

// ---------------- Hadamard(groups of 8) + Born measurement ----------------
__global__ void had_born_kernel(const float2* __restrict__ H,        // [N, SD] mapped h
                                const float* __restrict__ phi_raw,   // [N, 2*SD]
                                float* __restrict__ out) {            // [N, SD]
  size_t idx = (size_t)blockIdx.x * blockDim.x + threadIdx.x;
  if (idx >= (size_t)NROW * SD_) return;
  int n = (int)(idx / SD_), d = (int)(idx % SD_);
  int g = d / PP, q = d % PP;
  const float2* base = H + (size_t)n * SD_ + g * PP;
  const float scale = 0.35355339059327373f;           // 8^(-1/2)
  float hr = 0.f, hi = 0.f;
#pragma unroll
  for (int p = 0; p < PP; ++p) {
    float s = (__popc(p & q) & 1) ? -scale : scale;   // Sylvester Hadamard
    float2 v = base[p];
    hr += s * v.x; hi += s * v.y;
  }
  const size_t b2 = (size_t)n * 2 * SD_;
  float pre = phi_raw[b2 + d];
  float pim = phi_raw[b2 + SD_ + d];
  out[idx] = hr * pre + hi * pim;                     // Re(h * conj(phi))
}

// =============================== launch ===============================
extern "C" void kernel_launch(void* const* d_in, const int* in_sizes, int n_in,
                              void* d_out, int out_size, void* d_ws, size_t ws_size,
                              hipStream_t stream) {
  const float* x         = (const float*)d_in[0];
  const float* omega     = (const float*)d_in[1];
  const float* log_gamma = (const float*)d_in[2];
  const float* dt        = (const float*)d_in[3];
  const float* omega_s   = (const float*)d_in[4];
  const float* log_g_s   = (const float*)d_in[5];
  const float* dt_s      = (const float*)d_in[6];
  const float* W_psi     = (const float*)d_in[7];
  const float* b_psi     = (const float*)d_in[8];
  const float* W_phi     = (const float*)d_in[9];
  const float* b_phi     = (const float*)d_in[10];
  const float* W_psi_s   = (const float*)d_in[11];
  const float* b_psi_s   = (const float*)d_in[12];
  const float* W_B_s     = (const float*)d_in[13];
  const float* b_B_s     = (const float*)d_in[14];
  const float* W_B_tx    = (const float*)d_in[15];
  const float* b_B_tx    = (const float*)d_in[16];
  const float* W_B_th    = (const float*)d_in[17];
  const float* b_B_th    = (const float*)d_in[18];
  const float* W_ph_s    = (const float*)d_in[19];
  const float* b_ph_s    = (const float*)d_in[20];
  const float* W_ph_t    = (const float*)d_in[21];
  const float* b_ph_t    = (const float*)d_in[22];
  const float* W_gate    = (const float*)d_in[23];
  const float* b_gate    = (const float*)d_in[24];
  const float* tau       = (const float*)d_in[25];
  const float* beta      = (const float*)d_in[26];
  const float* fa_raw    = (const float*)d_in[27];
  const float* fb_real   = (const float*)d_in[28];
  const float* fb_imag   = (const float*)d_in[29];
  const float* fcp       = (const float*)d_in[30];
  const float* fdp       = (const float*)d_in[31];

  // ---- workspace carve-up (256B aligned) ----
  char* ws = (char*)d_ws;
  size_t off = 0;
  auto carve = [&](size_t bytes) -> char* {
    off = (off + 255) & ~(size_t)255;
    char* p = ws + off;
    off += bytes;
    return p;
  };
  bf16*   xb      = (bf16*)  carve((size_t)NROW * DD * sizeof(bf16));        // 16 MB
  bf16*   wscr    = (bf16*)  carve((size_t)2048 * 2048 * sizeof(bf16));      //  8 MB (max weight scratch)
  float*  gate    = (float*) carve((size_t)NROW * sizeof(float));
  float*  R1      = (float*) carve((size_t)NROW * 2 * SD_ * sizeof(float));  // psi_s_raw -> psi_raw
  float*  R2      = (float*) carve((size_t)NROW * 2 * SD_ * sizeof(float));  // ph_s_raw  -> ph_t_raw
  float*  R3      = (float*) carve((size_t)NROW * SC_ * sizeof(float));      // Bs_raw
  float*  R4      = (float*) carve((size_t)NROW * SD_ * sizeof(float));      // Bt_raw
  float*  R5      = (float*) carve((size_t)NROW * 2 * SD_ * sizeof(float));  // phi_raw
  float2* Ubuf    = (float2*)carve((size_t)NROW * SD_ * sizeof(float2));     // U_s -> U -> mapped h
  bf16*   feat    = (bf16*)  carve((size_t)NROW * 2 * SC_ * sizeof(bf16));   // [N,2SC] bf16
  (void)ws_size; (void)n_in; (void)in_sizes; (void)out_size;

  const int T = 256;
  auto cdiv = [](size_t a, size_t b) { return (unsigned)((a + b - 1) / b); };

  // pack x to bf16
  size_t nx = (size_t)NROW * DD;
  cvt_bf16_kernel<<<cdiv(nx, T), T, 0, stream>>>(x, xb, nx);

  // gate
  gate_kernel<<<cdiv(NROW, T), T, 0, stream>>>(x, W_gate, b_gate, tau, beta, gate);

  // generic "convert weight then GEMM" helper
  auto gemm = [&](const float* Wf, const float* bias, const float* addC, float* C,
                  const bf16* Abf, int N, int K) {
    size_t nw = (size_t)N * K;
    cvt_bf16_kernel<<<cdiv(nw, T), T, 0, stream>>>(Wf, wscr, nw);
    dim3 grid(N / BN, NROW / BM);
    wmma_gemm_bias<<<grid, 128, 0, stream>>>(Abf, wscr, bias, addC, C, NROW, N, K);
  };

  // ---- scout wave ----
  gemm(W_psi_s, b_psi_s, nullptr, R1, xb, 2 * SC_, DD);   // psi_s_raw
  gemm(W_ph_s,  b_ph_s,  nullptr, R2, xb, 2 * SC_, DD);   // ph_s_raw
  gemm(W_B_s,   b_B_s,   nullptr, R3, xb, SC_,     DD);   // Bs_raw
  size_t nu = (size_t)NROW * SC_;
  u_kernel<<<cdiv(nu, T), T, 0, stream>>>(R1, R2, R3, gate, Ubuf);
  scan_scout_kernel<<<cdiv(BB * SC_, T), T, 0, stream>>>(Ubuf, omega_s, log_g_s, dt_s, feat);

  // ---- true wave: Bt = x@W_B_tx + b_B_tx + feat@W_B_th + b_B_th ----
  gemm(W_B_tx, b_B_tx, nullptr, R4, xb,   SD_, DD);
  gemm(W_B_th, b_B_th, R4,      R4, feat, SD_, 2 * SC_);  // accumulate onto Bt_raw
  gemm(W_psi,  b_psi,  nullptr, R1, xb, 2 * SD_, DD);     // psi_raw (reuse R1)
  gemm(W_ph_t, b_ph_t, nullptr, R2, xb, 2 * SD_, DD);     // ph_t_raw (reuse R2)
  u_kernel<<<cdiv(nu, T), T, 0, stream>>>(R1, R2, R4, gate, Ubuf);
  scan_true_kernel<<<cdiv(BB * SD_, T), T, 0, stream>>>(Ubuf, omega, log_gamma, dt,
                                                        fa_raw, fb_real, fb_imag, fcp, fdp);

  // ---- measurement ----
  gemm(W_phi, b_phi, nullptr, R5, xb, 2 * SD_, DD);       // phi_raw
  had_born_kernel<<<cdiv(nu, T), T, 0, stream>>>(Ubuf, R5, (float*)d_out);
}
